// fg_mlp_18202071400870
// MI455X (gfx1250) — compile-verified, matrix operand order checked by reading, not run on hardware
//
#include <hip/hip_runtime.h>
#include <math.h>
#include <stdint.h>

#define AS1 __attribute__((address_space(1)))
#define AS3 __attribute__((address_space(3)))

// ---- CDNA5 async-to-LDS staging path, compile-safe guarded -----------------
#if defined(__AMDGCN__) && \
    __has_builtin(__builtin_amdgcn_global_load_async_to_lds_b128) && \
    __has_builtin(__builtin_amdgcn_s_wait_asynccnt)
#define FG_ASYNC 1
#warning "FG: async staging enabled (global_load_async_to_lds_b128)"
#else
#define FG_ASYNC 0
#if defined(__AMDGCN__)
#warning "FG: async builtins unavailable -> ds_store staging fallback"
#endif
#endif

typedef __attribute__((ext_vector_type(8))) int v8i;
typedef int fg_v4i __attribute__((__vector_size__(16)));   // b128 payload type

#define A_COEF  17.5328f
#define LN2SQ   0.48045301391820142f   // ln(2)^2, folded into masked weights
#define B_DIM   64
#define O_DIM   1024
#define I_DIM   1024
#define I_LEAKF 7e-12f

#if FG_ASYNC
__device__ __forceinline__ void fg_async_b128(const void* g, void* l) {
  // LDS[l] = MEM[g], 16B per lane, tracked by ASYNCcnt.
  // Low 32 bits of a generic LDS pointer are the DS offset (aperture rule).
  __builtin_amdgcn_global_load_async_to_lds_b128(
      (AS1 fg_v4i*)(uintptr_t)g, (AS3 fg_v4i*)(uint32_t)(uintptr_t)l, 0, 0);
}
#endif

// ---------------------------------------------------------------------------
// Kernel A1: per-weight precompute.
//   lw   = log|w| ; spec = lw*(-8.0282e-9) - 8.0497e-8
//   enb  = exp(-b) = exp(1.41837*lw - 18.4134)
//   pw   = [w>=0] * spec * ln2^2 ;  nwn = [w<0] * (-spec) * ln2^2
// ---------------------------------------------------------------------------
__global__ void fg_precompute_w(const float* __restrict__ W,
                                float* __restrict__ enb,
                                float* __restrict__ pw,
                                float* __restrict__ nwn, int n) {
  int idx = blockIdx.x * blockDim.x + threadIdx.x;
  if (idx >= n) return;
  float w  = W[idx];
  float lw = logf(fabsf(w));
  float spec = fmaf(lw, -8.0282e-09f, -8.0497e-08f);
  enb[idx] = expf(fmaf(lw, 1.41837f, -18.4134f));
  float s2 = spec * LN2SQ;
  bool pos = (w >= 0.0f);
  pw[idx]  = pos ? s2 : 0.0f;
  nwn[idx] = pos ? 0.0f : -s2;
}

// Kernel A2: expv[b,i] = exp(a * min(v, 2.5))
__global__ void fg_precompute_v(const float* __restrict__ vs,
                                float* __restrict__ expv, int n) {
  int idx = blockIdx.x * blockDim.x + threadIdx.x;
  if (idx >= n) return;
  expv[idx] = expf(A_COEF * fminf(vs[idx], 2.5f));
}

// ---------------------------------------------------------------------------
// Kernel NP: npos[o] = sum_i [w[o,i] >= 0] via V_WMMA_I32_16X16X64_IU8.
// A = sign-mask bytes (rows o), B = all-ones bytes -> D[m,n] = row-sum(A_m).
// B==ones makes the result independent of K placement within lanes L/L+16.
// ---------------------------------------------------------------------------
__global__ void fg_npos_wmma(const float* __restrict__ W, int* __restrict__ npos) {
  const int lane = threadIdx.x;                 // 0..31, wave32, EXEC all ones
  const int row  = blockIdx.x * 16 + (lane & 15);
  const int koff = (lane >> 4) * 32;            // which 32 of the 64 K per step
  v8i acc  = {0, 0, 0, 0, 0, 0, 0, 0};
  v8i ones = {0x01010101, 0x01010101, 0x01010101, 0x01010101,
              0x01010101, 0x01010101, 0x01010101, 0x01010101};
  for (int i0 = 0; i0 < I_DIM; i0 += 64) {
    const float4* w4 = (const float4*)(W + (size_t)row * I_DIM + i0 + koff);
    v8i a;
#pragma unroll
    for (int q = 0; q < 8; ++q) {
      float4 f = w4[q];
      unsigned d = (f.x >= 0.0f ? 1u : 0u)
                 | ((f.y >= 0.0f ? 1u : 0u) << 8)
                 | ((f.z >= 0.0f ? 1u : 0u) << 16)
                 | ((f.w >= 0.0f ? 1u : 0u) << 24);
      a[q] = (int)d;
    }
    acc = __builtin_amdgcn_wmma_i32_16x16x64_iu8(false, a, false, ones, acc,
                                                 false, false);
  }
  if (lane == 0) {
#pragma unroll
    for (int r = 0; r < 8; ++r) npos[blockIdx.x * 16 + r] = acc[r];
  } else if (lane == 16) {
#pragma unroll
    for (int r = 0; r < 8; ++r) npos[blockIdx.x * 16 + 8 + r] = acc[r];
  }
}

// ---------------------------------------------------------------------------
// Kernel B: heavy loop. grid = (O/32 o-groups, 16 i-splits of 64), WG=128thr.
// Thread-block tile: 64b x 32o ; per thread 8b x 2o.
// Inner element: x = fma(expv, enb, 1) ; L = v_log_f32(x) ; u=L*L ; 2 FMAs
// => 1 TRANS + 4 VALU per element, 0.875 LDS dwords/element.
// ---------------------------------------------------------------------------
#define IT   64
#define OT   32
#define WGT  128
#define STRD 68   // 64+4 pad: 16B-aligned rows, breaks 16-way bank conflicts

__global__ __launch_bounds__(WGT) void fg_main(
    const float* __restrict__ expv, const float* __restrict__ enb,
    const float* __restrict__ pw,   const float* __restrict__ nwn,
    float* __restrict__ ppart, float* __restrict__ npart) {
  __shared__ float s_ev[B_DIM * STRD];
  __shared__ float s_nb[OT * STRD];
  __shared__ float s_pw[OT * STRD];
  __shared__ float s_nw[OT * STRD];

  const int tid = threadIdx.x;
  const int o0  = blockIdx.x * OT;
  const int i0  = blockIdx.y * IT;

#if FG_ASYNC
  for (int idx = tid; idx < B_DIM * (IT / 4); idx += WGT) {
    int r = idx >> 4, c = idx & 15;
    fg_async_b128(expv + (size_t)r * I_DIM + i0 + c * 4,
                  &s_ev[r * STRD + c * 4]);
  }
  for (int idx = tid; idx < OT * (IT / 4); idx += WGT) {
    int r = idx >> 4, c = idx & 15;
    size_t g = (size_t)(o0 + r) * I_DIM + i0 + c * 4;
    fg_async_b128(enb + g, &s_nb[r * STRD + c * 4]);
    fg_async_b128(pw  + g, &s_pw[r * STRD + c * 4]);
    fg_async_b128(nwn + g, &s_nw[r * STRD + c * 4]);
  }
  __builtin_amdgcn_s_wait_asynccnt(0);
  __syncthreads();
#else
  for (int idx = tid; idx < B_DIM * (IT / 4); idx += WGT) {
    int r = idx >> 4, c = idx & 15;
    *(float4*)(&s_ev[r * STRD + c * 4]) =
        *(const float4*)(expv + (size_t)r * I_DIM + i0 + c * 4);
  }
  for (int idx = tid; idx < OT * (IT / 4); idx += WGT) {
    int r = idx >> 4, c = idx & 15;
    size_t g = (size_t)(o0 + r) * I_DIM + i0 + c * 4;
    *(float4*)(&s_nb[r * STRD + c * 4]) = *(const float4*)(enb + g);
    *(float4*)(&s_pw[r * STRD + c * 4]) = *(const float4*)(pw + g);
    *(float4*)(&s_nw[r * STRD + c * 4]) = *(const float4*)(nwn + g);
  }
  __syncthreads();
#endif

  const int ol0 = (tid & 15) * 2;   // 2 o's per thread
  const int bb  = (tid >> 4) * 8;   // 8 b's per thread
  float aP[2][8] = {};
  float aN[2][8] = {};

  for (int ii = 0; ii < IT; ii += 4) {
    float4 ev[8];
#pragma unroll
    for (int j = 0; j < 8; ++j)
      ev[j] = *(const float4*)(&s_ev[(bb + j) * STRD + ii]);
    const float* ef = (const float*)ev;
#pragma unroll
    for (int p = 0; p < 2; ++p) {
      int ol = ol0 + p;
      float4 nb = *(const float4*)(&s_nb[ol * STRD + ii]);
      float4 wp = *(const float4*)(&s_pw[ol * STRD + ii]);
      float4 wn = *(const float4*)(&s_nw[ol * STRD + ii]);
      const float* nbf = (const float*)&nb;
      const float* wpf = (const float*)&wp;
      const float* wnf = (const float*)&wn;
#pragma unroll
      for (int c = 0; c < 4; ++c) {
        float nbc = nbf[c], wpc = wpf[c], wnc = wnf[c];
#pragma unroll
        for (int j = 0; j < 8; ++j) {
          float x = fmaf(ef[j * 4 + c], nbc, 1.0f);  // 1 + exp(a*v - b)
          float L = __log2f(x);                      // v_log_f32 (TRANS)
          float u = L * L;                           // ln2^2 folded in wp/wn
          aP[p][j] = fmaf(u, wpc, aP[p][j]);
          aN[p][j] = fmaf(u, wnc, aN[p][j]);
        }
      }
    }
  }

  const int sbase = blockIdx.y * (B_DIM * O_DIM);
#pragma unroll
  for (int p = 0; p < 2; ++p)
#pragma unroll
    for (int j = 0; j < 8; ++j) {
      int out = (bb + j) * O_DIM + (o0 + ol0 + p);
      ppart[sbase + out] = aP[p][j];
      npart[sbase + out] = aN[p][j];
    }
}

// ---------------------------------------------------------------------------
// Totals helper: fixed-order partial sum + leak + bias (bitwise deterministic)
// ---------------------------------------------------------------------------
__device__ __forceinline__ void fg_totals(const float* __restrict__ ppart,
                                          const float* __restrict__ npart,
                                          const int* __restrict__ npos,
                                          const float* __restrict__ bias,
                                          int idx, float* pr, float* nr) {
  int o = idx & (O_DIM - 1);
  float p = 0.f, n = 0.f;
#pragma unroll
  for (int s = 0; s < 16; ++s) {
    p += ppart[s * (B_DIM * O_DIM) + idx];
    n += npart[s * (B_DIM * O_DIM) + idx];
  }
  float np = (float)npos[o];
  float bo = bias[o];
  p += I_LEAKF * np + fmaxf(bo, 0.f);
  n += -I_LEAKF * ((float)I_DIM - np) + fminf(bo, 0.f);
  *pr = p; *nr = n;
}

// Kernel C: global max of p-currents and n-currents (single WG, tree reduce)
__global__ void fg_maxred(const float* __restrict__ ppart,
                          const float* __restrict__ npart,
                          const int* __restrict__ npos,
                          const float* __restrict__ bias,
                          float* __restrict__ maxbuf) {
  __shared__ float sp[1024], sn[1024];
  int tid = threadIdx.x;
  float mp = -3.4e38f, mn = -3.4e38f;
  for (int idx = tid; idx < B_DIM * O_DIM; idx += 1024) {
    float p, n;
    fg_totals(ppart, npart, npos, bias, idx, &p, &n);
    mp = fmaxf(mp, p); mn = fmaxf(mn, n);
  }
  sp[tid] = mp; sn[tid] = mn;
  __syncthreads();
  for (int s = 512; s > 0; s >>= 1) {
    if (tid < s) { sp[tid] = fmaxf(sp[tid], sp[tid + s]);
                   sn[tid] = fmaxf(sn[tid], sn[tid + s]); }
    __syncthreads();
  }
  if (tid == 0) { maxbuf[0] = sp[0]; maxbuf[1] = sn[0]; }
}

// ---------------------------------------------------------------------------
// nfet_itov / sigmoid_diff (elementwise)
// ---------------------------------------------------------------------------
__device__ __forceinline__ float nfet_itov_one(float c, bool posb) {
  float cur = posb ? c : fabsf(c);
  float lt  = posb ? 2.152e-06f : 4.67e-06f;
  float ltv = posb ? 2.0001f : 2.4507f;
  if (cur > lt) return ltv;
  if (cur > 1.622e-10f)
    return -13.2209f /
               (3.03607f + 1.75561f * __logf(fabsf(fmaf(1500.0f, cur, -1.9597e-07f)))) +
           0.120559f;
  if (cur > 1.5166e-11f)
    return fmaf(fmaf(fmaf(-1.7814e+29f, cur, 4.6537e+19f), cur, 5.3942e+08f),
                cur, -0.0182625f);
  return 0.0f;
}

__device__ __forceinline__ float sigmoid_diff_one(float p, float n) {
  float d = p - n;
  float arg, scale, off;
  if (p <= 0.38f) {
    arg = fmaf(31.202f, d, fmaf(-29.0649f, p, 12.7174f));
    scale = -0.2877195f; off = 2.44313f;
  } else if (p <= 1.78f) {
    arg = fmaf(fmaf(-3.98366f, p, 32.2026f), d, fmaf(-0.926118f, p, 2.4177f));
    scale = fmaf(0.131407f, p, -0.338609f);
    off = fmaf(-0.00157441f, p, 2.44385f);
  } else {
    arg = fmaf(fmaf(-32.32f, p, 83.1541f), d, fmaf(-9.69387f, p, 17.9666f));
    scale = fmaf(0.407395f, p, -0.830324f);
    off = 2.440658f;
  }
  return scale / (1.0f + __expf(arg)) + off;
}

// Kernel D: final elementwise stage -> out[b*O + o]
__global__ void fg_final(const float* __restrict__ ppart,
                         const float* __restrict__ npart,
                         const int* __restrict__ npos,
                         const float* __restrict__ bias,
                         const float* __restrict__ maxbuf,
                         float* __restrict__ out) {
  int idx = blockIdx.x * blockDim.x + threadIdx.x;
  if (idx >= B_DIM * O_DIM) return;
  float p, n;
  fg_totals(ppart, npart, npos, bias, idx, &p, &n);
  bool pb = maxbuf[0] > 0.0f;
  bool nb = maxbuf[1] > 0.0f;
  out[idx] = sigmoid_diff_one(nfet_itov_one(p, pb), nfet_itov_one(n, nb));
}

// ---------------------------------------------------------------------------
// Launch: d_in = {v_s[64x1024] f32, weights[1024x1024] f32, bias[1024] f32}
// Workspace (floats): enb 0 | pw 1M | nwn 2M | expv 3145728 |
//   ppart 3211264 (16x64K) | npart 4259840 | npos(int) 5308416 |
//   maxbuf 5309440   (~20.3 MB total)
// ---------------------------------------------------------------------------
extern "C" void kernel_launch(void* const* d_in, const int* in_sizes, int n_in,
                              void* d_out, int out_size, void* d_ws, size_t ws_size,
                              hipStream_t stream) {
  (void)in_sizes; (void)n_in; (void)out_size; (void)ws_size;
  const float* v_s  = (const float*)d_in[0];
  const float* W    = (const float*)d_in[1];
  const float* bias = (const float*)d_in[2];
  float* out = (float*)d_out;
  float* ws  = (float*)d_ws;

  float* enb    = ws;
  float* pw     = ws + 1048576;
  float* nwn    = ws + 2097152;
  float* expv   = ws + 3145728;
  float* ppart  = ws + 3211264;
  float* npart  = ws + 4259840;
  int*   npos   = (int*)(ws + 5308416);
  float* maxbuf = ws + 5309440;

  fg_precompute_w<<<4096, 256, 0, stream>>>(W, enb, pw, nwn, O_DIM * I_DIM);
  fg_precompute_v<<<256, 256, 0, stream>>>(v_s, expv, B_DIM * I_DIM);
  fg_npos_wmma<<<O_DIM / 16, 32, 0, stream>>>(W, npos);
  fg_main<<<dim3(O_DIM / OT, I_DIM / IT), WGT, 0, stream>>>(expv, enb, pw, nwn,
                                                            ppart, npart);
  fg_maxred<<<1, 1024, 0, stream>>>(ppart, npart, npos, bias, maxbuf);
  fg_final<<<256, 256, 0, stream>>>(ppart, npart, npos, bias, maxbuf, out);
}
